// _DualGraphGAT_24670292148714
// MI455X (gfx1250) — matrix operand order repaired and checked by reading, not compiled, not run on hardware
//
#include <hip/hip_runtime.h>

typedef __bf16 bf16;
typedef __attribute__((ext_vector_type(16))) __bf16 v16bf;
typedef __attribute__((ext_vector_type(8)))  float  v8f;

struct alignas(16) BF8 { bf16 v[8]; };

// ---------------------------------------------------------------- utilities
__global__ void __launch_bounds__(256)
k_f32_to_bf16(const float* __restrict__ in, bf16* __restrict__ out, int n) {
    int i = blockIdx.x * 256 + threadIdx.x;
    if (i < n) out[i] = (bf16)in[i];
}

__global__ void __launch_bounds__(256)
k_fill(float* __restrict__ p, float v, int n) {
    int i = blockIdx.x * 256 + threadIdx.x;
    if (i < n) p[i] = v;
}

// ---------------------------------------------------------------- WMMA GEMM
// C[M,Nc] = A[M,K](bf16) * W[Nc,K]^T(bf16) (+bias) (+relu)
// one wave per 16x16 output tile; K multiple of 32; M,Nc multiples of 16.
__global__ void __launch_bounds__(256)
k_wmma_gemm(const bf16* __restrict__ A, const bf16* __restrict__ W,
            const float* __restrict__ bias, float* __restrict__ Cf,
            bf16* __restrict__ Cb, int M, int K, int Nc, int relu)
{
    const int lane = threadIdx.x & 31;
    const int wid  = blockIdx.x * 8 + (threadIdx.x >> 5);
    const int tN   = Nc >> 4;
    if (wid >= (M >> 4) * tN) return;            // whole-wave uniform: EXEC stays full
    const int tm = wid / tN, tn = wid % tN;
    const int mrow = tm << 4, ncol = tn << 4;

    // A: lane l -> row (l&15); K chunks [koff..koff+7] and [koff+16..koff+23], koff=(l>>4)*8
    const bf16* Ap = A + (size_t)(mrow + (lane & 15)) * K + ((lane >> 4) << 3);
    // B=W^T: lane l -> col (l&15); K chunks [koff..koff+7] and [koff+16..koff+23], koff=(l>>4)*8
    const bf16* Wp = W + (size_t)(ncol + (lane & 15)) * K + ((lane >> 4) << 3);

    v8f acc = {};
    for (int kb = 0; kb < K; kb += 32) {
        __builtin_prefetch(Ap + kb + 64, 0, 0);   // global_prefetch_b8
        __builtin_prefetch(Wp + kb + 64, 0, 0);
        BF8 a0 = *(const BF8*)(Ap + kb);
        BF8 a1 = *(const BF8*)(Ap + kb + 16);
        BF8 b0 = *(const BF8*)(Wp + kb);
        BF8 b1 = *(const BF8*)(Wp + kb + 16);
        v16bf av, bv;
        #pragma unroll
        for (int i = 0; i < 8; ++i) {
            av[i] = a0.v[i]; av[i + 8] = a1.v[i];
            bv[i] = b0.v[i]; bv[i + 8] = b1.v[i];
        }
        acc = __builtin_amdgcn_wmma_f32_16x16x32_bf16(
                  false, av, false, bv, (short)0, acc, false, false);
    }

    const int orow = mrow + ((lane >> 4) << 3);
    const int ocol = ncol + (lane & 15);
    const float bb = bias ? bias[ocol] : 0.f;
    #pragma unroll
    for (int v = 0; v < 8; ++v) {
        float r = acc[v] + bb;
        if (relu) r = fmaxf(r, 0.f);
        if (Cf) Cf[(size_t)(orow + v) * Nc + ocol] = r;
        if (Cb) Cb[(size_t)(orow + v) * Nc + ocol] = (bf16)r;
    }
}

// ---------------------------------------------------------------- attention
// per-node logits: esrc[n,h] = xh[n,h,:]·a_src[h,:]; one wave per node
__global__ void __launch_bounds__(256)
k_node_alpha(const float* __restrict__ xh, const float* __restrict__ a_src,
             const float* __restrict__ a_dst, float* __restrict__ esrc,
             float* __restrict__ edst, int Nn)
{
    const int lane = threadIdx.x & 31;
    const int n = blockIdx.x * 8 + (threadIdx.x >> 5);
    if (n >= Nn) return;
    const float* row = xh + (size_t)n * 256;
    #pragma unroll
    for (int h = 0; h < 4; ++h) {
        float x0 = row[h * 64 + lane], x1 = row[h * 64 + 32 + lane];
        float s = x0 * a_src[h * 64 + lane] + x1 * a_src[h * 64 + 32 + lane];
        float d = x0 * a_dst[h * 64 + lane] + x1 * a_dst[h * 64 + 32 + lane];
        #pragma unroll
        for (int off = 16; off; off >>= 1) {
            s += __shfl_xor(s, off, 32);
            d += __shfl_xor(d, off, 32);
        }
        if (lane == 0) { esrc[n * 4 + h] = s; edst[n * 4 + h] = d; }
    }
}

__device__ inline void atomicMaxF(float* addr, float val) {
    unsigned old = __float_as_uint(*addr);
    while (__uint_as_float(old) < val) {
        unsigned assumed = old;
        old = atomicCAS((unsigned*)addr, assumed, __float_as_uint(val));
        if (old == assumed) break;
    }
}

__device__ inline float leaky(float a) { return a > 0.f ? a : 0.2f * a; }

// phase 1: segment max of leaky(e_src[src]+e_dst[dst]) per (dst, head)
__global__ void __launch_bounds__(256)
k_edge_max(const int* __restrict__ eidx, const float* __restrict__ esrc,
           const float* __restrict__ edst, float* __restrict__ amax, int E, int Nn)
{
    int e = blockIdx.x * 256 + threadIdx.x;
    if (e >= E + Nn) return;
    int s, d;
    if (e < E) { s = eidx[e]; d = eidx[E + e]; } else { s = d = e - E; }
    #pragma unroll
    for (int h = 0; h < 4; ++h)
        atomicMaxF(&amax[d * 4 + h], leaky(esrc[s * 4 + h] + edst[d * 4 + h]));
}

// phase 2: ae = exp(a - amax[dst]); den += ae
__global__ void __launch_bounds__(256)
k_edge_exp(const int* __restrict__ eidx, const float* __restrict__ esrc,
           const float* __restrict__ edst, const float* __restrict__ amax,
           float* __restrict__ ae, float* __restrict__ den, int E, int Nn)
{
    int e = blockIdx.x * 256 + threadIdx.x;
    if (e >= E + Nn) return;
    int s, d;
    if (e < E) { s = eidx[e]; d = eidx[E + e]; } else { s = d = e - E; }
    #pragma unroll
    for (int h = 0; h < 4; ++h) {
        float a = leaky(esrc[s * 4 + h] + edst[d * 4 + h]);
        float v = __expf(a - amax[d * 4 + h]);
        ae[(size_t)e * 4 + h] = v;
        atomicAdd(&den[d * 4 + h], v);
    }
}

// phase 3: agg[dst,:] += ae[e,h(c)] * xh[src,:]; one wave per edge, coalesced lanes
__global__ void __launch_bounds__(256)
k_edge_agg(const int* __restrict__ eidx, const float* __restrict__ ae,
           const float* __restrict__ xh, float* __restrict__ agg, int E, int Nn)
{
    const int lane = threadIdx.x & 31;
    const int e = blockIdx.x * 8 + (threadIdx.x >> 5);
    if (e >= E + Nn) return;
    int s, d;
    if (e < E) { s = eidx[e]; d = eidx[E + e]; } else { s = d = e - E; }
    float w0 = ae[(size_t)e * 4 + 0], w1 = ae[(size_t)e * 4 + 1];
    float w2 = ae[(size_t)e * 4 + 2], w3 = ae[(size_t)e * 4 + 3];
    const float* xr = xh + (size_t)s * 256;
    float* ar = agg + (size_t)d * 256;
    #pragma unroll
    for (int j = 0; j < 8; ++j) {
        int c = lane + 32 * j;
        float w = (c < 64) ? w0 : (c < 128) ? w1 : (c < 192) ? w2 : w3;
        atomicAdd(&ar[c], xr[c] * w);
    }
}

// phase 4: out = agg / (den + 1e-16) + b
__global__ void __launch_bounds__(256)
k_node_fin(const float* __restrict__ agg, const float* __restrict__ den,
           const float* __restrict__ b, float* __restrict__ out, int Nn)
{
    int i = blockIdx.x * 256 + threadIdx.x;
    if (i >= Nn * 256) return;
    int n = i >> 8, c = i & 255;
    out[i] = agg[i] / (den[n * 4 + (c >> 6)] + 1e-16f) + b[c];
}

// gate + elu -> bf16 (layer-0 output)
__global__ void __launch_bounds__(256)
k_combine0(const float* __restrict__ hs, const float* __restrict__ he,
           const float* __restrict__ g, bf16* __restrict__ hb, int n)
{
    int i = blockIdx.x * 256 + threadIdx.x;
    if (i >= n) return;
    float gg = 1.f / (1.f + __expf(-g[0]));
    float v = gg * hs[i] + (1.f - gg) * he[i];
    v = v > 0.f ? v : (__expf(v) - 1.f);
    hb[i] = (bf16)v;
}

// gate only -> z (f32, to d_out) + bf16 mirror
__global__ void __launch_bounds__(256)
k_combine1(const float* __restrict__ hs, const float* __restrict__ he,
           const float* __restrict__ g, float* __restrict__ z,
           bf16* __restrict__ zb, int n)
{
    int i = blockIdx.x * 256 + threadIdx.x;
    if (i >= n) return;
    float gg = 1.f / (1.f + __expf(-g[0]));
    float v = gg * hs[i] + (1.f - gg) * he[i];
    z[i] = v;
    zb[i] = (bf16)v;
}

// ---------------------------------------------------------------- launcher
extern "C" void kernel_launch(void* const* d_in, const int* in_sizes, int n_in,
                              void* d_out, int out_size, void* d_ws, size_t ws_size,
                              hipStream_t stream)
{
    const int Nn = 20000, DIM = 256, OUTP = 64;
    const int E = in_sizes[1] / 2;
    const int Etot = E + Nn;

    const float* x       = (const float*)d_in[0];
    const int*   eidx_s  = (const int*)d_in[1];
    const int*   eidx_e  = (const int*)d_in[2];
    const float* W0[2]   = { (const float*)d_in[3],  (const float*)d_in[7]  };
    const float* as0[2]  = { (const float*)d_in[4],  (const float*)d_in[8]  };
    const float* ad0[2]  = { (const float*)d_in[5],  (const float*)d_in[9]  };
    const float* b0[2]   = { (const float*)d_in[6],  (const float*)d_in[10] };
    const float* g0      = (const float*)d_in[11];
    const float* W1[2]   = { (const float*)d_in[12], (const float*)d_in[16] };
    const float* as1[2]  = { (const float*)d_in[13], (const float*)d_in[17] };
    const float* ad1[2]  = { (const float*)d_in[14], (const float*)d_in[18] };
    const float* b1[2]   = { (const float*)d_in[15], (const float*)d_in[19] };
    const float* g1      = (const float*)d_in[20];
    const float* Wp1 = (const float*)d_in[21]; const float* bp1 = (const float*)d_in[22];
    const float* Wp2 = (const float*)d_in[23]; const float* bp2 = (const float*)d_in[24];
    const float* Wr1 = (const float*)d_in[25]; const float* br1 = (const float*)d_in[26];
    const float* Wr2 = (const float*)d_in[27]; const float* br2 = (const float*)d_in[28];

    // ---- workspace carve-out (256B aligned)
    char* cur = (char*)d_ws;
    auto alloc = [&](size_t bytes) -> void* {
        void* p = cur; cur += (bytes + 255) & ~(size_t)255; return p;
    };
    bf16* xb   = (bf16*)alloc((size_t)Nn * 256 * 2);
    bf16* hb   = (bf16*)alloc((size_t)Nn * 256 * 2);
    bf16* zb   = (bf16*)alloc((size_t)Nn * 256 * 2);
    bf16* t1b  = (bf16*)alloc((size_t)Nn * 64 * 2);
    bf16* t2b  = (bf16*)alloc((size_t)Nn * 64 * 2);
    bf16* Wb0[2] = { (bf16*)alloc(256*256*2), (bf16*)alloc(256*256*2) };
    bf16* Wb1[2] = { (bf16*)alloc(256*256*2), (bf16*)alloc(256*256*2) };
    bf16* Wp1b = (bf16*)alloc(64*256*2);
    bf16* Wp2b = (bf16*)alloc(64*64*2);
    bf16* Wr1b = (bf16*)alloc(64*256*2);
    bf16* Wr2b = (bf16*)alloc(256*64*2);
    float* xh   = (float*)alloc((size_t)Nn * 256 * 4);
    float* agg  = (float*)alloc((size_t)Nn * 256 * 4);
    float* hs   = (float*)alloc((size_t)Nn * 256 * 4);
    float* he   = (float*)alloc((size_t)Nn * 256 * 4);
    float* esrc = (float*)alloc((size_t)Nn * 4 * 4);
    float* edst = (float*)alloc((size_t)Nn * 4 * 4);
    float* amax = (float*)alloc((size_t)Nn * 4 * 4);
    float* den  = (float*)alloc((size_t)Nn * 4 * 4);
    float* ae   = (float*)alloc((size_t)Etot * 4 * 4);

    auto cdiv = [](int a, int b) { return (a + b - 1) / b; };
    auto cvt = [&](const float* src, bf16* dst, int n) {
        k_f32_to_bf16<<<cdiv(n, 256), 256, 0, stream>>>(src, dst, n);
    };
    auto gemm = [&](const bf16* A, const bf16* W, const float* bias,
                    float* Cf, bf16* Cb, int M, int K, int Nc, int relu) {
        int tiles = (M >> 4) * (Nc >> 4);
        k_wmma_gemm<<<cdiv(tiles, 8), 256, 0, stream>>>(A, W, bias, Cf, Cb, M, K, Nc, relu);
    };

    // ---- precision conversions (one-time per call)
    cvt(x, xb, Nn * 256);
    cvt(W0[0], Wb0[0], 256*256); cvt(W0[1], Wb0[1], 256*256);
    cvt(W1[0], Wb1[0], 256*256); cvt(W1[1], Wb1[1], 256*256);
    cvt(Wp1, Wp1b, 64*256); cvt(Wp2, Wp2b, 64*64);
    cvt(Wr1, Wr1b, 64*256); cvt(Wr2, Wr2b, 256*64);

    // ---- one GAT convolution (shared scratch, sequential on stream)
    auto gat = [&](const bf16* inB, const bf16* Wb, const float* a_s, const float* a_d,
                   const float* bias, const int* eidx, float* hout) {
        gemm(inB, Wb, nullptr, xh, nullptr, Nn, 256, DIM, 0);
        k_node_alpha<<<cdiv(Nn, 8), 256, 0, stream>>>(xh, a_s, a_d, esrc, edst, Nn);
        k_fill<<<cdiv(Nn * 4, 256), 256, 0, stream>>>(amax, -3e38f, Nn * 4);
        k_fill<<<cdiv(Nn * 4, 256), 256, 0, stream>>>(den, 0.f, Nn * 4);
        k_fill<<<cdiv(Nn * 256, 256), 256, 0, stream>>>(agg, 0.f, Nn * 256);
        k_edge_max<<<cdiv(Etot, 256), 256, 0, stream>>>(eidx, esrc, edst, amax, E, Nn);
        k_edge_exp<<<cdiv(Etot, 256), 256, 0, stream>>>(eidx, esrc, edst, amax, ae, den, E, Nn);
        k_edge_agg<<<cdiv(Etot, 8), 256, 0, stream>>>(eidx, ae, xh, agg, E, Nn);
        k_node_fin<<<cdiv(Nn * 256, 256), 256, 0, stream>>>(agg, den, bias, hout, Nn);
    };

    float* out   = (float*)d_out;
    float* prot  = out;                                   // [N,64]
    float* rna   = out + (size_t)Nn * OUTP;               // [N,256]
    float* zout  = out + (size_t)Nn * OUTP + (size_t)Nn * DIM;  // [N,256]

    // ---- layer 0
    gat(xb, Wb0[0], as0[0], ad0[0], b0[0], eidx_s, hs);
    gat(xb, Wb0[1], as0[1], ad0[1], b0[1], eidx_e, he);
    k_combine0<<<cdiv(Nn * 256, 256), 256, 0, stream>>>(hs, he, g0, hb, Nn * 256);

    // ---- layer 1
    gat(hb, Wb1[0], as1[0], ad1[0], b1[0], eidx_s, hs);
    gat(hb, Wb1[1], as1[1], ad1[1], b1[1], eidx_e, he);
    k_combine1<<<cdiv(Nn * 256, 256), 256, 0, stream>>>(hs, he, g1, zout, zb, Nn * 256);

    // ---- heads (WMMA)
    gemm(zb, Wp1b, bp1, nullptr, t1b, Nn, 256, 64, 1);
    gemm(t1b, Wp2b, bp2, prot, nullptr, Nn, 64, 64, 0);
    gemm(zb, Wr1b, br1, nullptr, t2b, Nn, 256, 64, 1);
    gemm(t2b, Wr2b, br2, rna, nullptr, Nn, 64, 256, 0);

    (void)n_in; (void)out_size; (void)ws_size;
}
